// DSLDSCell_37271726195424
// MI455X (gfx1250) — matrix operand order repaired and checked by reading, not compiled, not run on hardware
//
#include <hip/hip_runtime.h>
#include <hip/hip_bf16.h>

typedef __attribute__((ext_vector_type(16))) _Float16 v16h;
typedef __attribute__((ext_vector_type(4)))  _Float16 v4h;
typedef __attribute__((ext_vector_type(8)))  float    v8f;

#define NN 4096
#define DD 16
#define KK 64
#define XX 128
#define HH 256
#define LOG2PI_F 1.8378770664093453f

// ---------------- RNG (counter-based; JAX streams not reproducible, no runtime check) ----
__device__ inline unsigned pcg_hash(unsigned x) {
    x = x * 747796405u + 2891336453u;
    unsigned w = ((x >> ((x >> 28u) + 4u)) ^ x) * 277803737u;
    return (w >> 22u) ^ w;
}
__device__ inline float rng_uniform(unsigned idx, unsigned salt) {
    return ((pcg_hash(idx ^ salt) >> 8) + 0.5f) * (1.0f / 16777216.0f);
}
__device__ inline float rng_gumbel(unsigned idx, unsigned salt) {
    float u = rng_uniform(idx, salt);
    return -logf(-logf(u));
}
__device__ inline float rng_normal(unsigned idx, unsigned salt) {
    float u1 = rng_uniform(idx * 2u + 11u, salt);
    float u2 = rng_uniform(idx * 2u + 12345u, salt ^ 0x85ebca6bu);
    return sqrtf(-2.0f * logf(u1)) * cosf(6.283185307179586f * u2);
}
__device__ inline float geluf(float x) {
    float x3 = x * x * x;
    float t = tanhf(0.7978845608028654f * (x + 0.044715f * x3));
    return 0.5f * x * (1.0f + t);
}

// ---------------- pack weight f32 [Kreal x Nout] -> f16 WMMA-fragment order ------------
// Layout: dst[((nt*KT + kt)*512) + lane*16 + i]; a wave's 16x16x32 B fragment for tile
// (kt,nt) is 1KB contiguous (one v16h per lane). Zero-pads K to KT*32 and N to NT*16.
__global__ __launch_bounds__(256) void pack_bfrag_k(
    const float* __restrict__ src, _Float16* __restrict__ dst,
    int Kreal, int Nout, int KT, int NT)
{
    int idx = blockIdx.x * 256 + threadIdx.x;
    int total = KT * NT * 512;
    if (idx >= total) return;
    int tile = idx >> 9;
    int rem  = idx & 511;
    int lane = rem >> 4, ii = rem & 15;
    int kt = tile % KT, nt = tile / KT;
    int grp = lane >> 4, nn = lane & 15;
    int k = kt * 32 + grp * 16 + ii;
    int c = nt * 16 + nn;
    float v = (k < Kreal && c < Nout) ? src[(size_t)k * Nout + c] : 0.0f;
    dst[idx] = (_Float16)v;
}

// ---------------- pack [W(1024x128); c(64x128)] -> fragment order (Kd=1088) ------------
__global__ __launch_bounds__(256) void pack_wc_k(
    const float* __restrict__ W, const float* __restrict__ cmat, _Float16* __restrict__ dst)
{
    const int KT = 34;
    int idx = blockIdx.x * 256 + threadIdx.x;
    if (idx >= KT * 8 * 512) return;
    int tile = idx >> 9;
    int rem  = idx & 511;
    int lane = rem >> 4, ii = rem & 15;
    int kt = tile % KT, nt = tile / KT;
    int grp = lane >> 4, nn = lane & 15;
    int k = kt * 32 + grp * 16 + ii;
    int c = nt * 16 + nn;
    float v = (k < 1024) ? W[(size_t)k * 128 + c] : cmat[(size_t)(k - 1024) * 128 + c];
    dst[idx] = (_Float16)v;
}

__global__ void pad_bias_k(const float* __restrict__ src, float* __restrict__ dst, int nreal) {
    int i = threadIdx.x;
    dst[i] = (i < nreal) ? src[i] : 0.0f;
}

// ---------------- concat + where(isfinite) for fz input --------------------------------
__global__ __launch_bounds__(256) void concat_fz_k(
    const float* __restrict__ zs, const float* __restrict__ ks, const float* __restrict__ xt,
    float* __restrict__ zf, float* __restrict__ kf, float* __restrict__ raw)
{
    int n = blockIdx.x, t = threadIdx.x;
    for (int i = t; i < DD; i += 256) {
        float z = zs[n * DD + i];
        float v = __builtin_isfinite(z) ? z : 0.0f;
        zf[n * DD + i] = v;
        raw[n * 208 + i] = v;
    }
    for (int i = t; i < KK; i += 256) {
        float k = ks[n * KK + i];
        float v = __builtin_isfinite(k) ? k : 1.0f;
        kf[n * KK + i] = v;
        raw[n * 208 + DD + i] = v;
    }
    for (int i = t; i < XX; i += 256) {
        raw[n * 208 + DD + KK + i] = xt[n * XX + i];
    }
}

// ---------------- concat for ez input: [gt(128), k_new(64), xt(128)] -------------------
__global__ __launch_bounds__(256) void concat_ez_k(
    const float* __restrict__ gt, const float* __restrict__ knew, const float* __restrict__ xt,
    float* __restrict__ raw)
{
    int n = blockIdx.x, t = threadIdx.x;
    for (int i = t; i < 320; i += 256) {
        float v;
        if (i < 128)        v = gt[n * 128 + i];
        else if (i < 192)   v = knew[n * 64 + (i - 128)];
        else                v = xt[n * 128 + (i - 192)];
        raw[n * 320 + i] = v;
    }
}

// ---------------- row layernorm: f32 in -> f16 out (pads [L,ostride) with zeros) -------
__global__ __launch_bounds__(256) void layernorm_k(
    const float* __restrict__ in, int istride,
    const float* __restrict__ s, const float* __restrict__ b,
    _Float16* __restrict__ out, int ostride, int L)
{
    __shared__ float red[256];
    int n = blockIdx.x, t = threadIdx.x;
    const float* x = in + (size_t)n * istride;
    float ls = 0.f, lq = 0.f;
    for (int i = t; i < L; i += 256) { float v = x[i]; ls += v; lq += v * v; }
    red[t] = ls; __syncthreads();
    for (int st = 128; st > 0; st >>= 1) { if (t < st) red[t] += red[t + st]; __syncthreads(); }
    float mean = red[0] / (float)L; __syncthreads();
    red[t] = lq; __syncthreads();
    for (int st = 128; st > 0; st >>= 1) { if (t < st) red[t] += red[t + st]; __syncthreads(); }
    float var = red[0] / (float)L - mean * mean;
    float inv = rsqrtf(var + 1e-6f);
    for (int i = t; i < ostride; i += 256) {
        float o = 0.0f;
        if (i < L) o = (x[i] - mean) * inv * s[i] + b[i];
        out[(size_t)n * ostride + i] = (_Float16)o;
    }
}

// ---------------- WMMA GEMM: C[M,Npad] = act(A[M,Kd] @ B + bias) -----------------------
// Grid EXACTLY covers the (padded) output: no guards anywhere; EXEC stays all-ones.
// A: f16 row-major (K zero-padded). Bpk: fragment-packed f16 (K and N padded).
// Block = 128 thr = 4 waves; tile = 16 rows x 64 cols. A tile double-buffered in LDS
// (1 barrier/K-step, v4h staging); B fragments stream from L2 as 2x b128 per lane.
__global__ __launch_bounds__(128) void gemm_wmma_k(
    const _Float16* __restrict__ A, int lda,
    const _Float16* __restrict__ Bpk,
    const float* __restrict__ bias,
    float* __restrict__ C, int ldc,
    int Kd, int act)
{
    __shared__ alignas(32) _Float16 As[2][512];
    int tid  = threadIdx.x;
    int wave = tid >> 5;
    int lane = tid & 31;
    int row0 = blockIdx.x * 16;
    int wcol = blockIdx.y * 64 + wave * 16;
    int KT = Kd >> 5;

    // per-thread A staging map: 4 consecutive k stay within one fragment octet
    int i4 = tid * 4;
    int m = i4 >> 5, kbase = i4 & 31;
    int grp = kbase >> 4;
    int kp = kbase - 16 * grp;
    int idxb = kp;
    const _Float16* Arow = A + (size_t)(row0 + m) * lda + kbase;
    _Float16* Asl = &As[0][(grp * 16 + m) * 16 + idxb];

    const _Float16* Bw = Bpk + ((size_t)(blockIdx.y * 4 + wave) * KT) * 512 + lane * 16;

    v8f cacc = {0.f, 0.f, 0.f, 0.f, 0.f, 0.f, 0.f, 0.f};

    *(v4h*)Asl = *(const v4h*)Arow;              // prologue: stage tile 0 into buffer 0
#pragma unroll 2
    for (int kt = 0; kt < KT - 1; ++kt) {        // branchless steady state
        __syncthreads();
        *(v4h*)(Asl + ((kt + 1) & 1) * 512) = *(const v4h*)(Arow + (size_t)(kt + 1) * 32);
        const v16h afrag = *(const v16h*)(&As[kt & 1][lane * 16]);
        const v16h bfrag = *(const v16h*)(Bw + (size_t)kt * 512);
        __builtin_prefetch(Bw + (size_t)(kt + 2) * 512, 0, 1);  // speculative; OK to overrun
        cacc = __builtin_amdgcn_wmma_f32_16x16x32_f16(
            false, afrag, false, bfrag, (short)0, cacc, false, false);
    }
    __syncthreads();                              // epilogue iteration
    {
        const v16h afrag = *(const v16h*)(&As[(KT - 1) & 1][lane * 16]);
        const v16h bfrag = *(const v16h*)(Bw + (size_t)(KT - 1) * 512);
        cacc = __builtin_amdgcn_wmma_f32_16x16x32_f16(
            false, afrag, false, bfrag, (short)0, cacc, false, false);
    }

    int cg = lane >> 4;
    int ni = lane & 15;
#pragma unroll
    for (int j = 0; j < 8; ++j) {
        int r = row0 + j + cg * 8;                // C layout: VGPR j -> M = j + 8*laneGroup
        int cidx = wcol + ni;
        float v = cacc[j];
        if (bias) v += bias[cidx];
        if (act == 1) v = geluf(v);
        C[(size_t)r * ldc + cidx] = v;
    }
}

// ---------------- block reductions over 64 threads -------------------------------------
__device__ inline float redmax64(float v, float* red) {
    int t = threadIdx.x;
    red[t] = v; __syncthreads();
    for (int s = 32; s > 0; s >>= 1) { if (t < s) red[t] = fmaxf(red[t], red[t + s]); __syncthreads(); }
    float r = red[0]; __syncthreads();
    return r;
}
__device__ inline float redsum64(float v, float* red) {
    int t = threadIdx.x;
    red[t] = v; __syncthreads();
    for (int s = 32; s > 0; s >>= 1) { if (t < s) red[t] += red[t + s]; __syncthreads(); }
    float r = red[0]; __syncthreads();
    return r;
}

// ---------------- Gumbel-softmax sampling of k + Concrete logprob diff -----------------
__global__ __launch_bounds__(64) void sample_k_k(
    const float* __restrict__ qkl, const float* __restrict__ kf,
    const float* __restrict__ ksample, const float* __restrict__ transition,
    const float* __restrict__ start_pk, const unsigned char* __restrict__ mask,
    const float* __restrict__ temp_p,
    float* __restrict__ knew_out, float* __restrict__ diwae_out,
    float* __restrict__ out_qk_softmax)
{
    __shared__ float red[64];
    __shared__ float kfs[64];
    int n = blockIdx.x, j = threadIdx.x;
    float temp = temp_p[0];
    float ql = qkl[n * 64 + j];
    kfs[j] = kf[n * 64 + j];
    __syncthreads();

    float g = rng_gumbel((unsigned)(n * 64 + j), 0x13579bdu);

    float sq = (ql + g) / temp;
    float mq = redmax64(sq, red);
    float eq = expf(sq - mq);
    float qk_samp = eq / redsum64(eq, red);

    float dot = 0.f;
    const float* trn = transition + (size_t)n * 64 * 64;
    for (int k = 0; k < 64; ++k) dot += kfs[k] * trn[k * 64 + j];
    float pl = __builtin_isfinite(ksample[n * 64 + j]) ? logf(dot) : logf(start_pk[j]);
    float sp = (pl + g) / temp;
    float mp = redmax64(sp, red);
    float ep = expf(sp - mp);
    float pk_samp = ep / redsum64(ep, red);

    float kn = mask[n] ? qk_samp : pk_samp;
    knew_out[n * 64 + j] = kn;

    // d_iwae = clp(kn,ql) - clp(kn,pl); row-constant terms cancel
    float lx = logf(kn);
    float scq = ql - temp * lx;
    float scp = pl - temp * lx;
    float ssq = redsum64(scq, red);
    float ssp = redsum64(scp, red);
    float mxq = redmax64(scq, red);
    float lseq = mxq + logf(redsum64(expf(scq - mxq), red));
    float mxp = redmax64(scp, red);
    float lsep = mxp + logf(redsum64(expf(scp - mxp), red));
    if (j == 0) diwae_out[n] = (ssq - 64.f * lseq) - (ssp - 64.f * lsep);

    float mr = redmax64(ql, red);
    float er = expf(ql - mr);
    out_qk_softmax[n * 64 + j] = er / redsum64(er, red);
}

// ---------------- Ab/Q mixing (HBM-bound streaming), Cholesky, L^-1 columns ------------
__global__ __launch_bounds__(256) void pz_chol_k(
    const float* __restrict__ knew, const float* __restrict__ zf, const float* __restrict__ zs,
    const float* __restrict__ Ab, const float* __restrict__ Q,
    const float* __restrict__ start_mu, const float* __restrict__ start_Q,
    float* __restrict__ pzmu, float* __restrict__ Lout,
    float* __restrict__ sinv, float* __restrict__ ldet)
{
    __shared__ float kn[64];
    __shared__ float M[256];
    __shared__ float Aw[256];
    __shared__ float bw[16];
    int n = blockIdx.x, t = threadIdx.x;
    int d = t >> 4, e = t & 15;
    if (t < 64) kn[t] = knew[n * 64 + t];
    __syncthreads();

    const float* Abn = Ab + (size_t)n * 64 * 16 * 17;
    const float* Qn  = Q  + (size_t)n * 64 * 16 * 16;
    float accA = 0.f, accQ = 0.f;
    for (int k = 0; k < 64; ++k) {
        float kv = kn[k];
        accA += kv * Abn[(k * 16 + d) * 17 + e];    // A = Ab[..., :16]
        accQ += kv * Qn[(k * 16 + d) * 16 + e];
    }
    Aw[t] = accA;
    float zd = zs[n * 16 + d];
    M[t] = __builtin_isfinite(zd) ? accQ : start_Q[t];
    if (t < 16) {
        float s = 0.f;
        for (int k = 0; k < 64; ++k) s += kn[k] * Abn[(k * 16 + t) * 17 + 16]; // b = Ab[...,-1]
        bw[t] = s;
    }
    __syncthreads();

    if (t < 16) {
        float s = bw[t];
        for (int dd = 0; dd < 16; ++dd) s += zf[n * 16 + dd] * Aw[dd * 16 + t];
        float zv = zs[n * 16 + t];
        pzmu[n * 16 + t] = __builtin_isfinite(zv) ? s : start_mu[t];
    }

    if (t == 0) {  // serial 16x16 Cholesky in LDS
        float ld = 0.f;
        for (int j = 0; j < 16; ++j) {
            float s = M[j * 16 + j];
            for (int m = 0; m < j; ++m) s -= M[j * 16 + m] * M[j * 16 + m];
            float ljj = sqrtf(fmaxf(s, 1e-20f));
            M[j * 16 + j] = ljj;
            ld += 2.0f * logf(ljj);
            float inv = 1.0f / ljj;
            for (int i2 = j + 1; i2 < 16; ++i2) {
                float s2 = M[i2 * 16 + j];
                for (int m = 0; m < j; ++m) s2 -= M[i2 * 16 + m] * M[j * 16 + m];
                M[i2 * 16 + j] = s2 * inv;
            }
        }
        ldet[n] = ld;
    }
    __syncthreads();

    Lout[(size_t)n * 256 + t] = (e <= d) ? M[t] : 0.0f;

    if (t < 16) {   // column t of L^-1; sinv_diag[c] = sum_i Li[i,c]^2
        int c = t;
        float x[16];
        float ssum = 0.f;
        for (int i = 0; i < 16; ++i) {
            float s = (i == c) ? 1.0f : 0.0f;
            for (int m = c; m < i; ++m) s -= M[i * 16 + m] * x[m];
            x[i] = (i >= c) ? s / M[i * 16 + i] : 0.0f;
            ssum += x[i] * x[i];
        }
        sinv[n * 16 + c] = ssum;
    }
}

// ---------------- U16[n,:1024] = kron(k_new, zf); U16[n,1024:1088] = k_new (f16) -------
__global__ __launch_bounds__(256) void build_u_k(
    const float* __restrict__ knew, const float* __restrict__ zf, _Float16* __restrict__ U)
{
    int i = blockIdx.x * 256 + threadIdx.x;   // N*1088 threads
    int n = i / 1088;
    int r = i - n * 1088;
    float v;
    if (r < 1024) v = knew[n * 64 + (r >> 4)] * zf[n * 16 + (r & 15)];
    else          v = knew[n * 64 + (r - 1024)];
    U[i] = (_Float16)v;
}

__global__ void init_acc_k(float* acc) { if (threadIdx.x == 0 && blockIdx.x == 0) acc[0] = 0.f; }

// ---------------- per-row finalize: solves, sampling, KL, logprobs ---------------------
__global__ __launch_bounds__(256) void final_k(
    const float* __restrict__ ez, const float* __restrict__ pzmu, const float* __restrict__ Lm,
    const float* __restrict__ sinv, const float* __restrict__ ldet,
    const float* __restrict__ qzs, const unsigned char* __restrict__ mask,
    float* __restrict__ out_sample, float* __restrict__ klbuf, float* __restrict__ acc)
{
    int n = blockIdx.x * 256 + threadIdx.x;
    if (n >= NN) return;
    const float* Ln = Lm + (size_t)n * 256;

    float qsig[16], qmu[16], pmu[16];
#pragma unroll
    for (int i = 0; i < 16; ++i) {
        qsig[i] = expf(fmaxf(qzs[i], -3.0f));
        qmu[i]  = ez[n * 64 + i];              // padded ldc=64; split(...,2)[0]
        pmu[i]  = pzmu[n * 16 + i];
    }

    float y[16], sumy2 = 0.f, sumsq = 0.f, ldq = 0.f;
#pragma unroll
    for (int i = 0; i < 16; ++i) {
        float s = qmu[i] - pmu[i];
#pragma unroll
        for (int m = 0; m < i; ++m) s -= Ln[i * 16 + m] * y[m];
        y[i] = s / Ln[i * 16 + i];
        sumy2 += y[i] * y[i];
        sumsq += sinv[n * 16 + i] * qsig[i] * qsig[i];
        ldq += 2.0f * logf(qsig[i]);
    }
    float ldp = ldet[n];
    klbuf[n] = 0.5f * (sumsq + sumy2 - 16.0f + ldp - ldq);

    float eps[16], smp[16];
#pragma unroll
    for (int i = 0; i < 16; ++i) eps[i] = rng_normal((unsigned)(n * 16 + i), 0x2468aceu);
    unsigned char mv = mask[n];
#pragma unroll
    for (int i = 0; i < 16; ++i) {
        float le = 0.f;
#pragma unroll
        for (int m = 0; m <= i; ++m) le += Ln[i * 16 + m] * eps[m];
        float s1 = qmu[i] + qsig[i] * eps[i];
        float s2 = fminf(fmaxf(pmu[i] + le, -100.0f), 100.0f);
        smp[i] = mv ? s1 : s2;
        out_sample[n * 16 + i] = smp[i];
    }

    float qlp = 0.f;
#pragma unroll
    for (int i = 0; i < 16; ++i) {
        float zz = (smp[i] - qmu[i]) / qsig[i];
        qlp += -0.5f * zz * zz - logf(qsig[i]);
    }
    qlp -= 0.5f * 16.0f * LOG2PI_F;

    float w[16], sumw2 = 0.f;
#pragma unroll
    for (int i = 0; i < 16; ++i) {
        float s = smp[i] - pmu[i];
#pragma unroll
        for (int m = 0; m < i; ++m) s -= Ln[i * 16 + m] * w[m];
        w[i] = s / Ln[i * 16 + i];
        sumw2 += w[i] * w[i];
    }
    float plp = -0.5f * sumw2 - 0.5f * ldp - 0.5f * 16.0f * LOG2PI_F;

    atomicAdd(acc, qlp - plp);   // reference: iwae = (qz_lp - pz_lp).sum(-1) is a SCALAR
}

__global__ __launch_bounds__(256) void finalize_out_k(
    const float* __restrict__ klbuf, const float* __restrict__ diwae,
    const float* __restrict__ acc, float* __restrict__ out)
{
    int n = blockIdx.x * 256 + threadIdx.x;
    if (n < NN) {
        out[65536 + n] = klbuf[n] + diwae[n];     // kl + d_kl
        out[69632 + n] = acc[0] + diwae[n];       // scalar iwae + d_iwae
    }
}

// ======================================================================================
extern "C" void kernel_launch(void* const* d_in, const int* in_sizes, int n_in,
                              void* d_out, int out_size, void* d_ws, size_t ws_size,
                              hipStream_t stream)
{
    const float* temp       = (const float*)d_in[0];
    const float* z_sample   = (const float*)d_in[1];
    const float* k_sample   = (const float*)d_in[2];
    const float* xt         = (const float*)d_in[3];
    const unsigned char* mk = (const unsigned char*)d_in[4];
    const float* transition = (const float*)d_in[5];
    const float* start_pk   = (const float*)d_in[6];
    const float* Ab         = (const float*)d_in[7];
    const float* Q          = (const float*)d_in[8];
    const float* start_mu   = (const float*)d_in[9];
    const float* start_Q    = (const float*)d_in[10];
    const float* fz_ln0_s = (const float*)d_in[11];
    const float* fz_ln0_b = (const float*)d_in[12];
    const float* fz_w0    = (const float*)d_in[13];
    const float* fz_b0    = (const float*)d_in[14];
    const float* fz_ln1_s = (const float*)d_in[15];
    const float* fz_ln1_b = (const float*)d_in[16];
    const float* fz_w1    = (const float*)d_in[17];
    const float* fz_b1    = (const float*)d_in[18];
    const float* fz_ln2_s = (const float*)d_in[19];
    const float* fz_ln2_b = (const float*)d_in[20];
    const float* fz_w2    = (const float*)d_in[21];
    const float* fz_b2    = (const float*)d_in[22];
    const float* ez_ln0_s = (const float*)d_in[23];
    const float* ez_ln0_b = (const float*)d_in[24];
    const float* ez_w0    = (const float*)d_in[25];
    const float* ez_b0    = (const float*)d_in[26];
    const float* ez_ln1_s = (const float*)d_in[27];
    const float* ez_ln1_b = (const float*)d_in[28];
    const float* ez_w1    = (const float*)d_in[29];
    const float* ez_b1    = (const float*)d_in[30];
    const float* ez_ln2_s = (const float*)d_in[31];
    const float* ez_ln2_b = (const float*)d_in[32];
    const float* ez_w2    = (const float*)d_in[33];
    const float* ez_b2    = (const float*)d_in[34];
    const float* Wk       = (const float*)d_in[35];   // [K,D,X] == [1024 x 128]
    const float* ck       = (const float*)d_in[36];   // [64 x 128]
    const float* qz_sigma = (const float*)d_in[37];

    float* out = (float*)d_out;
    float* ws  = (float*)d_ws;

    // -------- workspace layout (float offsets; all even -> 8B aligned) --------
    float* zf   = ws + 0;            // N*16
    float* kf   = ws + 65536;        // N*64            -> 327680
    float* raw  = ws + 327680;       // N*320 (fz uses 208, ez uses 320) -> 1638400
    float* big  = ws + 1638400;      // N*1024 f32: hiddens; overlaid by U16 (f16)
    float* q0 = big;
    float* q2 = big + 2 * NN * 256;
    float* qk   = ws + 5832704;      // N*64  -> 6094848
    float* knew = ws + 6094848;      // N*64  -> 6356992
    float* diw  = ws + 6356992;      // N     -> 6361088
    float* pzmu = ws + 6361088;      // N*16  -> 6426624
    float* Lm   = ws + 6426624;      // N*256 -> 7475200
    float* sinv = ws + 7475200;      // N*16  -> 7540736
    float* ldet = ws + 7540736;      // N     -> 7544832
    float* gt   = ws + 7544832;      // N*128 -> 8069120
    float* ez   = ws + 8069120;      // N*64 (padded ldc) -> 8331264
    float* klb  = ws + 8331264;      // N     -> 8335360
    float* accv = ws + 8335360;      // 1 (pad 64) -> 8335424
    float* b2pad= ws + 8335424;      // 64 (padded ez_b2) -> 8335488
    // f16 regions
    _Float16* h_ln  = (_Float16*)(ws + 8335488);  // N*320 halves
    _Float16* h_lnA = (_Float16*)(ws + 8990848);  // N*256 halves
    _Float16* h_lnB = (_Float16*)(ws + 9252992);  // N*256 halves
    _Float16* wpack = (_Float16*)(ws + 9515136);  // 442368 halves, 32B aligned
    _Float16* fzw0_16 = wpack;                    // KT=7,  NT=16 : 57344
    _Float16* fzw1_16 = wpack + 57344;            // KT=8,  NT=16 : 65536
    _Float16* fzw2_16 = wpack + 122880;           // KT=8,  NT=4  : 16384
    _Float16* ezw0_16 = wpack + 139264;           // KT=10, NT=16 : 81920
    _Float16* ezw1_16 = wpack + 221184;           // KT=8,  NT=16 : 65536
    _Float16* ezw2_16 = wpack + 286720;           // KT=8,  NT=4 (N padded 32->64) : 16384
    _Float16* wc_16   = wpack + 303104;           // KT=34, NT=8  : 139264
    _Float16* U16 = (_Float16*)big;               // N*1088 halves (fz hiddens dead then)

    // -------- pack weights into WMMA-fragment order f16 (once per call; ~0.9 MB) ------
    pack_bfrag_k<<<224, 256, 0, stream>>>(fz_w0, fzw0_16, 208, 256, 7, 16);
    pack_bfrag_k<<<256, 256, 0, stream>>>(fz_w1, fzw1_16, 256, 256, 8, 16);
    pack_bfrag_k<<<64,  256, 0, stream>>>(fz_w2, fzw2_16, 256, 64,  8, 4);
    pack_bfrag_k<<<320, 256, 0, stream>>>(ez_w0, ezw0_16, 320, 256, 10, 16);
    pack_bfrag_k<<<256, 256, 0, stream>>>(ez_w1, ezw1_16, 256, 256, 8, 16);
    pack_bfrag_k<<<64,  256, 0, stream>>>(ez_w2, ezw2_16, 256, 32,  8, 4);
    pack_wc_k<<<544, 256, 0, stream>>>(Wk, ck, wc_16);
    pad_bias_k<<<1, 64, 0, stream>>>(ez_b2, b2pad, 32);

    // -------- fz net --------
    concat_fz_k<<<NN, 256, 0, stream>>>(z_sample, k_sample, xt, zf, kf, raw);
    layernorm_k<<<NN, 256, 0, stream>>>(raw, 208, fz_ln0_s, fz_ln0_b, h_ln, 224, 208);
    gemm_wmma_k<<<dim3(NN / 16, 4), 128, 0, stream>>>(h_ln, 224, fzw0_16, fz_b0, q0, 256, 224, 1);
    layernorm_k<<<NN, 256, 0, stream>>>(q0, 256, fz_ln1_s, fz_ln1_b, h_lnA, 256, 256);
    gemm_wmma_k<<<dim3(NN / 16, 4), 128, 0, stream>>>(h_lnA, 256, fzw1_16, fz_b1, q2, 256, 256, 1);
    layernorm_k<<<NN, 256, 0, stream>>>(q2, 256, fz_ln2_s, fz_ln2_b, h_lnB, 256, 256);
    gemm_wmma_k<<<dim3(NN / 16, 1), 128, 0, stream>>>(h_lnB, 256, fzw2_16, fz_b2, qk, 64, 256, 0);

    // -------- gumbel-softmax k sampling + concrete logprob diff + output 4 --------
    sample_k_k<<<NN, 64, 0, stream>>>(qk, kf, k_sample, transition, start_pk, mk, temp,
                                      knew, diw, out + 73728);

    // -------- Ab/Q mixtures + Cholesky (HBM-bound streaming: ~0.55 GB) --------
    pz_chol_k<<<NN, 256, 0, stream>>>(knew, zf, z_sample, Ab, Q, start_mu, start_Q,
                                      pzmu, Lm, sinv, ldet);

    // -------- gt = [kron(k_new, zf), k_new] @ [W ; c]  (single WMMA GEMM, Kd=1088) ----
    build_u_k<<<(NN * 1088) / 256, 256, 0, stream>>>(knew, zf, U16);
    gemm_wmma_k<<<dim3(NN / 16, 2), 128, 0, stream>>>(U16, 1088, wc_16, (const float*)nullptr, gt, 128, 1088, 0);

    // -------- ez net (hidden f32 buffers reuse `big`; U16 dead after gt GEMM) --------
    concat_ez_k<<<NN, 256, 0, stream>>>(gt, knew, xt, raw);
    layernorm_k<<<NN, 256, 0, stream>>>(raw, 320, ez_ln0_s, ez_ln0_b, h_ln, 320, 320);
    gemm_wmma_k<<<dim3(NN / 16, 4), 128, 0, stream>>>(h_ln, 320, ezw0_16, ez_b0, q0, 256, 320, 1);
    layernorm_k<<<NN, 256, 0, stream>>>(q0, 256, ez_ln1_s, ez_ln1_b, h_lnA, 256, 256);
    gemm_wmma_k<<<dim3(NN / 16, 4), 128, 0, stream>>>(h_lnA, 256, ezw1_16, ez_b1, q2, 256, 256, 1);
    layernorm_k<<<NN, 256, 0, stream>>>(q2, 256, ez_ln2_s, ez_ln2_b, h_lnB, 256, 256);
    gemm_wmma_k<<<dim3(NN / 16, 1), 128, 0, stream>>>(h_lnB, 256, ezw2_16, b2pad, ez, 64, 256, 0);

    // -------- finalize --------
    init_acc_k<<<1, 1, 0, stream>>>(accv);
    final_k<<<NN / 256, 256, 0, stream>>>(ez, pzmu, Lm, sinv, ldet, qz_sigma, mk,
                                          out, klb, accv);
    finalize_out_k<<<NN / 256, 256, 0, stream>>>(klb, diw, accv, out);
}
